// K_regression_67327907332576
// MI455X (gfx1250) — matrix operand order repaired and checked by reading, not compiled, not run on hardware
//
#include <hip/hip_runtime.h>
#include <math.h>

// ---------------------------------------------------------------------------
// Bidirectional Mamba (2 blocks) + final dense head for MI455X (gfx1250).
// GEMMs run on v_wmma_f32_16x16x32_f16 (f16 in, f32 accumulate), staged
// through LDS with vectorized, branch-free f32->f16 conversion
// (global_load_b128 -> cvt -> ds_store_b128). Conv + selective scan are VALU
// kernels parallelized over (batch, channel). wave32 everywhere.
// ---------------------------------------------------------------------------

#define D_MODEL   128
#define D_STATE   16
#define D_CONV    4
#define D_INNER   256
#define DT_RANK   8
#define SEQ_LEN   199
#define BATCH     256
#define M_TOTAL   (BATCH * SEQ_LEN)       // 50944 = 796 * 64

typedef __attribute__((ext_vector_type(16))) _Float16 v16h;
typedef __attribute__((ext_vector_type(8)))  _Float16 v8h;
typedef __attribute__((ext_vector_type(8)))  float    v8f;
typedef __attribute__((ext_vector_type(4)))  float    f32x4;

enum { EPI_NONE = 0, EPI_SOFTPLUS = 1, EPI_GELU = 2, EPI_RESID = 3 };

#define TILE_M 64
#define TILE_N 64
#define TILE_K 32
#define LDS_STRIDE 40   // 32 halves + 8 pad (80B rows, 16B-aligned) for banks

__device__ __forceinline__ void pack_store(_Float16* dst, f32x4 a, f32x4 b,
                                           f32x4 c, f32x4 d)
{
  v8h h0, h1;
#pragma unroll
  for (int i = 0; i < 4; ++i) {
    h0[i]     = (_Float16)a[i];
    h0[4 + i] = (_Float16)b[i];
    h1[i]     = (_Float16)c[i];
    h1[4 + i] = (_Float16)d[i];
  }
  *(v8h*)(dst)     = h0;   // ds_store_b128
  *(v8h*)(dst + 8) = h1;   // ds_store_b128
}

// ---------------------------------------------------------------------------
// Generic GEMM: C[M,N] = epi(A[M,K] * W[N,K]^T + bias (+ resid)), f32 I/O,
// f16 WMMA math. Block = 128 threads (4 waves); block tile 64x64; each wave
// owns a 16(M) x 64(N) strip = 4 accumulators of 16x16.
// FASTK=true  : K % 32 == 0 and lda % 4 == 0 -> unguarded vector staging.
// FASTK=false : clamped, branch-free scalar staging (dt_proj, K=8 only).
// ---------------------------------------------------------------------------
template <bool FASTK>
__global__ __launch_bounds__(128)
void gemm_f16wmma(const float* __restrict__ A, int lda,
                  const float* __restrict__ W,          // (N,K) row-major
                  const float* __restrict__ bias,       // may be null
                  const float* __restrict__ resid,      // used when epi==RESID
                  float* __restrict__ C, int ldc,
                  int M, int N, int K, int epi)
{
  __shared__ _Float16 sA[TILE_M * LDS_STRIDE];
  __shared__ _Float16 sB[TILE_N * LDS_STRIDE];

  const int tid  = threadIdx.x;
  const int wave = tid >> 5;
  const int lane = tid & 31;
  const int m0   = blockIdx.x * TILE_M;
  const int n0   = blockIdx.y * TILE_N;

  v8f acc[4];
#pragma unroll
  for (int j = 0; j < 4; ++j)
#pragma unroll
    for (int e = 0; e < 8; ++e) acc[j][e] = 0.0f;

  const int lrow  = tid >> 1;          // 0..63 : tile row loaded by this thread
  const int lcol0 = (tid & 1) * 16;    // 0 or 16 : first of 16 contiguous cols

  const int  gr     = m0 + lrow;                 // always < M (M % 64 == 0)
  const int  gn     = n0 + lrow;
  const bool bvalid = (gn < N);
  const int  gnc    = bvalid ? gn : (N - 1);     // clamped: loads always legal
  const float* arow = A + (size_t)gr * lda;
  const float* wrow = W + (size_t)gnc * K;
  _Float16* sa = &sA[lrow * LDS_STRIDE + lcol0];
  _Float16* sb = &sB[lrow * LDS_STRIDE + lcol0];

  const int half = lane >> 4;          // 0/1
  const int r16  = lane & 15;

  for (int k0 = 0; k0 < K; k0 += TILE_K) {
    if (FASTK) {
      // ---- A tile: 16 contiguous f32 as 4x b128, no guards ---------------
      const float* ap = arow + k0 + lcol0;
      f32x4 a0 = *(const f32x4*)(ap + 0);
      f32x4 a1 = *(const f32x4*)(ap + 4);
      f32x4 a2 = *(const f32x4*)(ap + 8);
      f32x4 a3 = *(const f32x4*)(ap + 12);
      // ---- B tile: row of W, clamped pointer + uniform zero-select -------
      const float* wp = wrow + k0 + lcol0;
      f32x4 b0 = *(const f32x4*)(wp + 0);
      f32x4 b1 = *(const f32x4*)(wp + 4);
      f32x4 b2 = *(const f32x4*)(wp + 8);
      f32x4 b3 = *(const f32x4*)(wp + 12);
      if (!bvalid) {
#pragma unroll
        for (int i = 0; i < 4; ++i) { b0[i] = 0.f; b1[i] = 0.f; b2[i] = 0.f; b3[i] = 0.f; }
      }
      pack_store(sa, a0, a1, a2, a3);
      pack_store(sb, b0, b1, b2, b3);
    } else {
      // ---- branch-free clamped scalar staging (tiny K only) --------------
#pragma unroll
      for (int i = 0; i < 16; ++i) {
        const int  k    = k0 + lcol0 + i;
        const bool kin  = (k < K);
        const int  kc   = kin ? k : (K - 1);
        float av = arow[kc];
        float wv = wrow[kc];
        if (!kin) { av = 0.f; wv = 0.f; }
        if (!bvalid) wv = 0.f;
        sa[i] = (_Float16)av;
        sb[i] = (_Float16)wv;
      }
    }
    __syncthreads();

    // ---- WMMA compute ----------------------------------------------------
    {
      // A fragment: row M = wave*16 + r16; elements 0..15 = K c0..c0+15
      // (c0 = half*16).
      const int c0 = half * 16;
      const _Float16* pa = &sA[(wave * 16 + r16) * LDS_STRIDE];
      v16h afrag;
#pragma unroll
      for (int i = 0; i < 8; ++i) {
        afrag[i]     = pa[c0 + i];
        afrag[8 + i] = pa[c0 + 8 + i];
      }
      const int kb = half * 16;  // B fragment: N = r16, K = half*16 + i
#pragma unroll
      for (int j = 0; j < 4; ++j) {
        const _Float16* pb = &sB[(j * 16 + r16) * LDS_STRIDE + kb];
        v16h bfrag;
#pragma unroll
        for (int i = 0; i < 16; ++i) bfrag[i] = pb[i];
        acc[j] = __builtin_amdgcn_wmma_f32_16x16x32_f16(
            /*neg_a=*/false, afrag, /*neg_b=*/false, bfrag,
            /*c_mod=*/(short)0, acc[j], /*reuse_a=*/false, /*reuse_b=*/false);
      }
    }
    __syncthreads();
  }

  // ---- epilogue + store (C/D layout: row = wave*16 + half*8 + v, col=r16) --
#pragma unroll
  for (int j = 0; j < 4; ++j) {
    const int col = n0 + j * 16 + r16;
    if (col >= N) continue;
    const float bv = bias ? bias[col] : 0.0f;
#pragma unroll
    for (int v = 0; v < 8; ++v) {
      const int row = m0 + wave * 16 + half * 8 + v;
      float x = acc[j][v] + bv;
      if (epi == EPI_SOFTPLUS) {
        x = (x > 20.0f) ? x : log1pf(__expf(x));
      } else if (epi == EPI_GELU) {
        x = 0.5f * x * (1.0f + erff(x * 0.70710678118654752f));
      } else if (epi == EPI_RESID) {
        x += resid[(size_t)row * ldc + col];
      }
      C[(size_t)row * ldc + col] = x;
    }
  }
}

// ---------------------------------------------------------------------------
// LayerNorm over D_MODEL=128 per row; one wave32 per row (4 elems/lane).
// Optional fused add of a second tensor (fwd + bwd before norm2).
// ---------------------------------------------------------------------------
__global__ __launch_bounds__(256)
void layernorm_add(const float* __restrict__ in1, const float* __restrict__ in2,
                   const float* __restrict__ wvec, const float* __restrict__ bvec,
                   float* __restrict__ out)
{
  const int wave = threadIdx.x >> 5;
  const int lane = threadIdx.x & 31;
  const int row  = blockIdx.x * 8 + wave;            // grid is exact

  const float* p1 = in1 + (size_t)row * D_MODEL;
  float vals[4];
  float s = 0.0f, sq = 0.0f;
#pragma unroll
  for (int i = 0; i < 4; ++i) {
    const int c = lane + 32 * i;
    float v = p1[c];
    if (in2) v += in2[(size_t)row * D_MODEL + c];
    vals[i] = v;
    s += v;
    sq += v * v;
  }
#pragma unroll
  for (int off = 16; off > 0; off >>= 1) {
    s  += __shfl_xor(s,  off, 32);
    sq += __shfl_xor(sq, off, 32);
  }
  const float mean = s * (1.0f / D_MODEL);
  const float var  = sq * (1.0f / D_MODEL) - mean * mean;
  const float inv  = rsqrtf(var + 1e-5f);
#pragma unroll
  for (int i = 0; i < 4; ++i) {
    const int c = lane + 32 * i;
    out[(size_t)row * D_MODEL + c] = (vals[i] - mean) * inv * wvec[c] + bvec[c];
  }
}

// ---------------------------------------------------------------------------
// Depthwise causal conv (k=4) + bias + SiLU. xc = xz[:, :256] (row stride 512).
// reverse=1 runs the time-flipped direction in-place (anti-causal, kernel
// flipped), so no reversed copy is ever materialized.
// ---------------------------------------------------------------------------
__global__ __launch_bounds__(256)
void conv_silu(const float* __restrict__ xz, const float* __restrict__ cw,
               const float* __restrict__ cb, float* __restrict__ u, int reverse)
{
  const int d  = threadIdx.x;               // channel 0..255
  const int bl = blockIdx.x;                // 0..M-1
  const int b  = bl / SEQ_LEN;
  const int l  = bl % SEQ_LEN;

  float acc = cb[d];
#pragma unroll
  for (int j = 0; j < D_CONV; ++j) {
    const int li = reverse ? (l + (D_CONV - 1) - j) : (l - (D_CONV - 1) + j);
    if (li >= 0 && li < SEQ_LEN)            // uniform per block
      acc += cw[d * D_CONV + j] * xz[((size_t)(b * SEQ_LEN + li)) * 512 + d];
  }
  u[(size_t)bl * 256 + d] = acc / (1.0f + __expf(-acc));   // silu
}

// ---------------------------------------------------------------------------
// Selective scan, fused with skip (+u*D) and gate (*silu(z)).
// Block = one batch (256 threads = all channels); 16-wide state in registers.
// B/C (per (b,l), shared by all channels) broadcast through LDS.
// ---------------------------------------------------------------------------
__global__ __launch_bounds__(256)
void selective_scan(const float* __restrict__ u, const float* __restrict__ delta,
                    const float* __restrict__ xdbl,   // (M,40) = dt | B | C
                    const float* __restrict__ xz,     // z at col 256, stride 512
                    const float* __restrict__ A_log, const float* __restrict__ Dp,
                    float* __restrict__ y, int reverse)
{
  const int b = blockIdx.x;
  const int d = threadIdx.x;
  __shared__ float sB[D_STATE], sC[D_STATE];

  float Arow[D_STATE], h[D_STATE];
#pragma unroll
  for (int s = 0; s < D_STATE; ++s) {
    Arow[s] = -__expf(A_log[d * D_STATE + s]);
    h[s] = 0.0f;
  }
  const float Dd = Dp[d];

  for (int t = 0; t < SEQ_LEN; ++t) {
    const int    l = reverse ? (SEQ_LEN - 1 - t) : t;
    const size_t r = (size_t)b * SEQ_LEN + l;
    if (d < D_STATE)            sB[d]           = xdbl[r * 40 + DT_RANK + d];
    else if (d < 2 * D_STATE)   sC[d - D_STATE] = xdbl[r * 40 + DT_RANK + D_STATE + (d - D_STATE)];
    __syncthreads();

    const float dt = delta[r * 256 + d];
    const float ut = u[r * 256 + d];
    const float du = dt * ut;
    float yv = 0.0f;
#pragma unroll
    for (int s = 0; s < D_STATE; ++s) {
      const float dA = __expf(dt * Arow[s]);
      h[s] = dA * h[s] + du * sB[s];
      yv += h[s] * sC[s];
    }
    const float zt = xz[r * 512 + 256 + d];
    yv += ut * Dd;
    yv *= zt / (1.0f + __expf(-zt));
    y[r * 256 + d] = yv;
    __syncthreads();
  }
}

// ---------------------------------------------------------------------------
// Final head: out[b,:] = post(h[b].flatten() @ dense_w^T + dense_b), N=2.
// Block per batch, LDS tree reduction.
// ---------------------------------------------------------------------------
__global__ __launch_bounds__(256)
void final_dense(const float* __restrict__ h, const float* __restrict__ wd,
                 const float* __restrict__ bd, float* __restrict__ out)
{
  const int K = SEQ_LEN * D_MODEL;                 // 25472
  const int b = blockIdx.x, tid = threadIdx.x;
  const float* hp = h + (size_t)b * K;

  float a0 = 0.0f, a1 = 0.0f;
  for (int k = tid; k < K; k += 256) {
    const float v = hp[k];
    a0 += v * wd[k];
    a1 += v * wd[K + k];
  }
  __shared__ float r0[256], r1[256];
  r0[tid] = a0; r1[tid] = a1;
  __syncthreads();
  for (int s = 128; s > 0; s >>= 1) {
    if (tid < s) { r0[tid] += r0[tid + s]; r1[tid] += r1[tid + s]; }
    __syncthreads();
  }
  if (tid == 0) {
    float o0 = r0[0] + bd[0], o1 = r1[0] + bd[1];
    o0 = fminf(fmaxf(o0, 0.0f), 1e12f);
    o1 = fminf(fmaxf(o1, 0.0f), 1e12f);
    if (o0 < 1e-7f) o0 = 0.0f;
    if (o1 < 1e-7f) o1 = 0.0f;
    out[b * 2 + 0] = o0;
    out[b * 2 + 1] = o1;
  }
}

// ---------------------------------------------------------------------------
// Host-side orchestration
// ---------------------------------------------------------------------------
struct MambaP {
  const float *A_log, *D, *conv_b, *conv_w, *dt_proj_b, *dt_proj_w,
              *in_proj_w, *out_proj_w, *x_proj_w;
};
struct BlockP {
  MambaP fwd, bwd;
  const float *norm1_w, *norm1_b, *norm2_w, *norm2_b,
              *ff_w1, *ff_b1, *ff_w2, *ff_b2;
};

extern "C" void kernel_launch(void* const* d_in, const int* in_sizes, int n_in,
                              void* d_out, int out_size, void* d_ws, size_t ws_size,
                              hipStream_t stream)
{
  (void)n_in; (void)out_size; (void)ws_size;
  const int X_ELEMS = BATCH * SEQ_LEN * D_MODEL;   // 6,520,832 — unique size

  const float* x = nullptr;
  const float *dense_w = nullptr, *dense_b = nullptr;
  BlockP blks[2];

  int idx = 0;
  auto P = [&](int i) { return (const float*)d_in[i]; };

  if (in_sizes[0] == X_ELEMS) {
    // dict insertion order: x, then params {blocks[{fwd,bwd,norms,ff}], dense_w, dense_b}
    x = P(idx++);
    for (int bi = 0; bi < 2; ++bi) {
      auto rdM = [&](MambaP& m) {
        m.in_proj_w = P(idx++); m.conv_w = P(idx++); m.conv_b = P(idx++);
        m.x_proj_w = P(idx++);  m.dt_proj_w = P(idx++); m.dt_proj_b = P(idx++);
        m.A_log = P(idx++);     m.D = P(idx++);          m.out_proj_w = P(idx++);
      };
      rdM(blks[bi].fwd);
      rdM(blks[bi].bwd);
      blks[bi].norm1_w = P(idx++); blks[bi].norm1_b = P(idx++);
      blks[bi].norm2_w = P(idx++); blks[bi].norm2_b = P(idx++);
      blks[bi].ff_w1 = P(idx++);   blks[bi].ff_b1 = P(idx++);
      blks[bi].ff_w2 = P(idx++);   blks[bi].ff_b2 = P(idx++);
    }
    dense_w = P(idx++); dense_b = P(idx++);
  } else {
    // JAX pytree sorted-key order: params first ('p' < 'x'), dicts key-sorted.
    for (int bi = 0; bi < 2; ++bi) {
      auto rdM = [&](MambaP& m) {
        m.A_log = P(idx++);     m.D = P(idx++);
        m.conv_b = P(idx++);    m.conv_w = P(idx++);
        m.dt_proj_b = P(idx++); m.dt_proj_w = P(idx++);
        m.in_proj_w = P(idx++); m.out_proj_w = P(idx++); m.x_proj_w = P(idx++);
      };
      rdM(blks[bi].bwd);                               // 'bwd' < 'ff*' < 'fwd'
      blks[bi].ff_b1 = P(idx++); blks[bi].ff_b2 = P(idx++);
      blks[bi].ff_w1 = P(idx++); blks[bi].ff_w2 = P(idx++);
      rdM(blks[bi].fwd);
      blks[bi].norm1_b = P(idx++); blks[bi].norm1_w = P(idx++);
      blks[bi].norm2_b = P(idx++); blks[bi].norm2_w = P(idx++);
    }
    dense_b = P(idx++); dense_w = P(idx++);
    x = P(idx++);
  }

  // ---- workspace layout (f32) --------------------------------------------
  const size_t M = (size_t)M_TOTAL;
  float* w = (float*)d_ws;
  float* f_h     = w; w += M * 128;   // block output (also residual source)
  float* f_xnorm = w; w += M * 128;   // norm1 output; reused as norm2 output
  float* f_xz    = w; w += M * 512;   // in_proj output; reused as ff1 output
  float* f_u     = w; w += M * 256;
  float* f_xdbl  = w; w += M * 40;
  float* f_delta = w; w += M * 256;
  float* f_y     = w; w += M * 256;
  float* f_fwd   = w; w += M * 128;
  float* f_bwd   = w; w += M * 128;

  auto gemm = [&](const float* A, int lda, const float* W, int N, int K,
                  const float* bias, const float* resid, float* C, int ldc, int epi) {
    dim3 grid(M_TOTAL / TILE_M, (N + TILE_N - 1) / TILE_N);
    if ((K % TILE_K) == 0 && (lda % 4) == 0) {
      gemm_f16wmma<true><<<grid, 128, 0, stream>>>(A, lda, W, bias, resid, C, ldc,
                                                   M_TOTAL, N, K, epi);
    } else {
      gemm_f16wmma<false><<<grid, 128, 0, stream>>>(A, lda, W, bias, resid, C, ldc,
                                                    M_TOTAL, N, K, epi);
    }
  };

  auto run_mamba = [&](const MambaP& mp, int reverse, float* outbuf) {
    // in_proj: (M,128) x (512,128)^T -> xz (M,512)
    gemm(f_xnorm, 128, mp.in_proj_w, 2 * D_INNER, D_MODEL, nullptr, nullptr,
         f_xz, 512, EPI_NONE);
    // depthwise conv + silu -> u (M,256)
    conv_silu<<<M_TOTAL, 256, 0, stream>>>(f_xz, mp.conv_w, mp.conv_b, f_u, reverse);
    // x_proj: (M,256) x (40,256)^T -> x_dbl (M,40)
    gemm(f_u, 256, mp.x_proj_w, DT_RANK + 2 * D_STATE, D_INNER, nullptr, nullptr,
         f_xdbl, 40, EPI_NONE);
    // dt_proj + softplus: (M,8) x (256,8)^T -> delta (M,256)  [slow-K path]
    gemm(f_xdbl, 40, mp.dt_proj_w, D_INNER, DT_RANK, mp.dt_proj_b, nullptr,
         f_delta, 256, EPI_SOFTPLUS);
    // selective scan + skip + gate -> y (M,256)
    selective_scan<<<BATCH, 256, 0, stream>>>(f_u, f_delta, f_xdbl, f_xz,
                                              mp.A_log, mp.D, f_y, reverse);
    // out_proj: (M,256) x (128,256)^T -> out (M,128)
    gemm(f_y, 256, mp.out_proj_w, D_MODEL, D_INNER, nullptr, nullptr,
         outbuf, 128, EPI_NONE);
  };

  const float* cur = x;
  for (int bi = 0; bi < 2; ++bi) {
    const BlockP& bp = blks[bi];
    // norm1
    layernorm_add<<<M_TOTAL / 8, 256, 0, stream>>>(cur, nullptr,
                                                   bp.norm1_w, bp.norm1_b, f_xnorm);
    run_mamba(bp.fwd, 0, f_fwd);
    run_mamba(bp.bwd, 1, f_bwd);
    // norm2(fwd + bwd) -> m (reuse f_xnorm)
    layernorm_add<<<M_TOTAL / 8, 256, 0, stream>>>(f_fwd, f_bwd,
                                                   bp.norm2_w, bp.norm2_b, f_xnorm);
    // ff1 + gelu -> reuse f_xz
    gemm(f_xnorm, 128, bp.ff_w1, 4 * D_MODEL, D_MODEL, bp.ff_b1, nullptr,
         f_xz, 4 * D_MODEL, EPI_GELU);
    // ff2 + bias + residual -> f_h (elementwise in-place-safe vs resid)
    gemm(f_xz, 4 * D_MODEL, bp.ff_w2, D_MODEL, 4 * D_MODEL, bp.ff_b2, cur,
         f_h, D_MODEL, EPI_RESID);
    cur = f_h;
  }

  final_dense<<<BATCH, 256, 0, stream>>>(cur, dense_w, dense_b, (float*)d_out);
}
